// Model_8933531975941
// MI455X (gfx1250) — compile-verified, matrix operand order checked by reading, not run on hardware
//
#include <hip/hip_runtime.h>
#include <hip/hip_bf16.h>

// ---------------- sizes ----------------
#define S   4096
#define D   300
#define HH  300     // H
#define HU  150
#define G4  600     // 4*HU
#define KP  320     // padded K (multiple of 32) for bf16 WMMA
#define NP1 1200    // N of GEMM1 (Wf|Wb concat)
#define NP2 304     // padded N of GEMM2 (W_p)

typedef __bf16 bf16;
typedef __attribute__((ext_vector_type(16))) __bf16 v16bf;
typedef __attribute__((ext_vector_type(8)))  float  v8f;

// ---------------- WMMA bf16 GEMM: C[M,N] = A[M,KP] * B^T[N,KP] ----------------
__device__ __forceinline__ v16bf load_frag_bf16(const bf16* rowBase, int k0, int half) {
  v16bf f;
  const bf16* p0 = rowBase + k0 + half * 8;        // elements 0..7  -> K = k0 + half*8 + w
  const bf16* p1 = rowBase + k0 + 16 + half * 8;   // elements 8..15 -> K = k0 + 16 + half*8 + w
#pragma unroll
  for (int w = 0; w < 8; ++w) { f[w] = p0[w]; f[8 + w] = p1[w]; }
  return f;
}

__global__ void wmma_gemm_bf16(const bf16* __restrict__ A,   // M x KP row-major
                               const bf16* __restrict__ BT,  // N x KP row-major (B transposed)
                               float* __restrict__ C,        // M x ldc
                               int ldc) {
  const int tileN = blockIdx.x;
  const int tileM = blockIdx.y;
  const int lane  = threadIdx.x;        // 32 threads = 1 wave per tile
  const int r     = lane & 15;
  const int half  = lane >> 4;

  const bf16* aRow = A  + (size_t)(tileM * 16 + r) * KP;
  const bf16* bRow = BT + (size_t)(tileN * 16 + r) * KP;

  v8f acc = {};
#pragma unroll
  for (int k0 = 0; k0 < KP; k0 += 32) {
    v16bf a = load_frag_bf16(aRow, k0, half);
    v16bf b = load_frag_bf16(bRow, k0, half);
    acc = __builtin_amdgcn_wmma_f32_16x16x32_bf16(false, a, false, b,
                                                  (short)0, acc, false, false);
  }
#pragma unroll
  for (int i = 0; i < 8; ++i) {
    int row = tileM * 16 + half * 8 + i;
    int col = tileN * 16 + r;
    C[(size_t)row * ldc + col] = acc[i];
  }
}

// ---------------- prep kernels ----------------
__global__ void build_emb_kernel(const int* __restrict__ sentence,
                                 const float* __restrict__ embMat,
                                 bf16* __restrict__ embB) {
  int s = blockIdx.x, k = threadIdx.x;               // 320 threads
  float v = (k < D) ? embMat[(size_t)sentence[s] * D + k] : 0.f;
  embB[(size_t)s * KP + k] = (bf16)v;
}

__global__ void build_wcat_kernel(const float* __restrict__ Wf,
                                  const float* __restrict__ Wb,
                                  bf16* __restrict__ WT) {
  int i = blockIdx.x * blockDim.x + threadIdx.x;     // NP1*KP elements
  if (i >= NP1 * KP) return;
  int n = i / KP, k = i % KP;
  float v = 0.f;
  if (k < D) v = (n < G4) ? Wf[(size_t)k * G4 + n] : Wb[(size_t)k * G4 + (n - G4)];
  WT[i] = (bf16)v;
}

__global__ void build_wp_kernel(const float* __restrict__ Wp, bf16* __restrict__ WT) {
  int i = blockIdx.x * blockDim.x + threadIdx.x;     // NP2*KP elements
  if (i >= NP2 * KP) return;
  int n = i / KP, k = i % KP;
  float v = (k < HH && n < HH) ? Wp[(size_t)k * HH + n] : 0.f;
  WT[i] = (bf16)v;
}

__global__ void zero_bf16_kernel(bf16* __restrict__ p, int n) {
  int i = blockIdx.x * blockDim.x + threadIdx.x;
  if (i < n) p[i] = (bf16)0.f;
}

// ---------------- sequential LSTM scan (one block per direction) ----------------
__device__ __forceinline__ float sigmoidf_(float x) { return 1.f / (1.f + __expf(-x)); }

__global__ void lstm_scan_kernel(const float* __restrict__ Z,     // S x 1200 (emb @ [Wf|Wb])
                                 const float* __restrict__ bF,
                                 const float* __restrict__ bB,
                                 const float* __restrict__ Uf,    // 150 x 600
                                 const float* __restrict__ Ub,
                                 float* __restrict__ hidden,      // S x 300 f32
                                 bf16*  __restrict__ hiddenB) {   // S x 320 bf16 (padded)
  const int dir = blockIdx.x;                 // 0 = fwd, 1 = bwd
  const float* b = dir ? bB : bF;
  const float* U = dir ? Ub : Uf;
  const int tid = threadIdx.x;                // 640 threads (20 waves)

  __shared__ float h_sh[HU];
  __shared__ float z_sh[G4];
  float c = 0.f;
  if (tid < HU) h_sh[tid] = 0.f;
  __syncthreads();

  for (int t = 0; t < S; ++t) {
    const int zrow = dir ? (S - 1 - t) : t;
    if (tid < G4) {
      float z = Z[(size_t)zrow * NP1 + dir * G4 + tid] + b[tid];
#pragma unroll 5
      for (int k = 0; k < HU; ++k) z += h_sh[k] * U[(size_t)k * G4 + tid];
      z_sh[tid] = z;
    }
    __syncthreads();
    if (tid < HU) {
      float zi = z_sh[tid], zf = z_sh[tid + HU], zg = z_sh[tid + 2 * HU], zo = z_sh[tid + 3 * HU];
      float ig = sigmoidf_(zi), fg = sigmoidf_(zf), og = sigmoidf_(zo);
      c = fg * c + ig * tanhf(zg);
      float h = og * tanhf(c);
      int pos = dir ? (S - 1 - t) : t;
      hidden [(size_t)pos * HH + dir * HU + tid] = h;
      hiddenB[(size_t)pos * KP + dir * HU + tid] = (bf16)h;
      h_sh[tid] = h;
    }
    __syncthreads();
  }
}

// ---------------- primary + secondary attention (one block per position) ----------------
__global__ void primary_attn_kernel(const float* __restrict__ outP,   // S x 304 (hidden@Wp, no bias)
                                    const float* __restrict__ bP,
                                    const float* __restrict__ hidden, // S x 300
                                    const int*   __restrict__ synIdx, // S x 4
                                    const float* __restrict__ embMat,
                                    const float* __restrict__ Ws,     // 600
                                    const float* __restrict__ bS,
                                    float* __restrict__ ch) {         // S x 600 = coeff*h_hat
  const int s = blockIdx.x;
  const int tid = threadIdx.x;  // 128 threads
  __shared__ float sOut[HH];
  __shared__ float sSyn[4][HH];
  __shared__ float sScore[4];
  __shared__ float sHH[2 * HH];
  __shared__ float red[128];
  __shared__ float sCoef;

  for (int d = tid; d < HH; d += 128) sOut[d] = outP[(size_t)s * NP2 + d] + bP[d];
#pragma unroll
  for (int k = 0; k < 4; ++k) {
    int idx = synIdx[s * 4 + k];
    for (int d = tid; d < HH; d += 128) sSyn[k][d] = embMat[(size_t)idx * D + d];
  }
  __syncthreads();

  const int wave = tid >> 5, lane = tid & 31;
  if (wave < 4) {
    float p = 0.f;
    for (int d = lane; d < HH; d += 32) p += sOut[d] * sSyn[wave][d];
#pragma unroll
    for (int off = 16; off > 0; off >>= 1) p += __shfl_down(p, off, 32);
    if (lane == 0) sScore[wave] = __expf(p);
  }
  __syncthreads();

  for (int d = tid; d < HH; d += 128) {
    float m = sScore[0] * sSyn[0][d] + sScore[1] * sSyn[1][d] +
              sScore[2] * sSyn[2][d] + sScore[3] * sSyn[3][d];
    sHH[d]      = hidden[(size_t)s * HH + d];
    sHH[HH + d] = m;
  }
  __syncthreads();

  float p = 0.f;
  for (int j = tid; j < 2 * HH; j += 128) p += sHH[j] * Ws[j];
  red[tid] = p;
  __syncthreads();
#pragma unroll
  for (int off = 64; off > 0; off >>= 1) {
    if (tid < off) red[tid] += red[tid + off];
    __syncthreads();
  }
  if (tid == 0) sCoef = __expf(tanhf(red[0] + bS[0]));
  __syncthreads();

  const float cf = sCoef;
  for (int j = tid; j < 2 * HH; j += 128) ch[(size_t)s * (2 * HH) + j] = cf * sHH[j];
}

// ---------------- deterministic two-stage reduction + heads ----------------
#define CHUNKS 32
#define ROWS_PER_CHUNK (S / CHUNKS)   // 128

__global__ void reduce_stage1_kernel(const float* __restrict__ ch, float* __restrict__ partial) {
  const int b = blockIdx.x;           // CHUNKS blocks
  const int j = threadIdx.x;          // 640 threads, first 600 active
  if (j >= 2 * HH) return;
  float acc = 0.f;
  const int s0 = b * ROWS_PER_CHUNK;
  for (int s = s0; s < s0 + ROWS_PER_CHUNK; ++s) acc += ch[(size_t)s * (2 * HH) + j];
  partial[b * (2 * HH) + j] = acc;
}

__global__ void final_heads_kernel(const float* __restrict__ partial,
                                   const float* __restrict__ Wemo,  // 600 x 8
                                   const float* __restrict__ bEmo,
                                   const float* __restrict__ Wsent, // 600 x 1
                                   const float* __restrict__ bSent,
                                   float* __restrict__ out) {       // 9 floats
  __shared__ float Hh[2 * HH];
  const int tid = threadIdx.x;        // 320 threads (10 waves)
  for (int j = tid; j < 2 * HH; j += 320) {
    float a = 0.f;
#pragma unroll
    for (int b = 0; b < CHUNKS; ++b) a += partial[b * (2 * HH) + j];
    Hh[j] = a;
  }
  __syncthreads();
  const int wave = tid >> 5, lane = tid & 31;
  if (wave < 9) {
    const float* W; int ld, col; float bias;
    if (wave < 8) { W = Wemo; ld = 8; col = wave; bias = bEmo[wave]; }
    else          { W = Wsent; ld = 1; col = 0;   bias = bSent[0]; }
    float p = 0.f;
    for (int j = lane; j < 2 * HH; j += 32) p += Hh[j] * W[(size_t)j * ld + col];
#pragma unroll
    for (int off = 16; off > 0; off >>= 1) p += __shfl_down(p, off, 32);
    if (lane == 0) out[wave] = p + bias;
  }
}

// ---------------- launcher ----------------
static inline size_t alignup(size_t x) { return (x + 255) & ~(size_t)255; }

extern "C" void kernel_launch(void* const* d_in, const int* in_sizes, int n_in,
                              void* d_out, int out_size, void* d_ws, size_t ws_size,
                              hipStream_t stream) {
  (void)in_sizes; (void)n_in; (void)out_size; (void)ws_size;

  const int*   sentence = (const int*)  d_in[0];
  const int*   synIdx   = (const int*)  d_in[1];
  const float* embMat   = (const float*)d_in[2];
  const float* W_f      = (const float*)d_in[3];
  const float* U_f      = (const float*)d_in[4];
  const float* b_f      = (const float*)d_in[5];
  const float* W_b      = (const float*)d_in[6];
  const float* U_b      = (const float*)d_in[7];
  const float* b_b      = (const float*)d_in[8];
  const float* W_p      = (const float*)d_in[9];
  const float* b_p      = (const float*)d_in[10];
  const float* W_s      = (const float*)d_in[11];
  const float* b_s      = (const float*)d_in[12];
  const float* W_emo    = (const float*)d_in[13];
  const float* b_emo    = (const float*)d_in[14];
  const float* W_sent   = (const float*)d_in[15];
  const float* b_sent   = (const float*)d_in[16];
  float* out = (float*)d_out;

  // workspace layout
  char* ws = (char*)d_ws;
  size_t off = 0;
  bf16*  embB    = (bf16*) (ws + off); off = alignup(off + (size_t)S * KP * sizeof(bf16));
  bf16*  WcatT   = (bf16*) (ws + off); off = alignup(off + (size_t)NP1 * KP * sizeof(bf16));
  bf16*  WpT     = (bf16*) (ws + off); off = alignup(off + (size_t)NP2 * KP * sizeof(bf16));
  float* Z       = (float*)(ws + off); off = alignup(off + (size_t)S * NP1 * sizeof(float));
  float* hidden  = (float*)(ws + off); off = alignup(off + (size_t)S * HH * sizeof(float));
  bf16*  hiddenB = (bf16*) (ws + off); off = alignup(off + (size_t)S * KP * sizeof(bf16));
  float* outP    = (float*)(ws + off); off = alignup(off + (size_t)S * NP2 * sizeof(float));
  float* ch      = (float*)(ws + off); off = alignup(off + (size_t)S * 2 * HH * sizeof(float));
  float* partial = (float*)(ws + off); off = alignup(off + (size_t)CHUNKS * 2 * HH * sizeof(float));

  // 1) prep: gather+cast embeddings, transpose+cast weights, zero padded hidden_bf16
  build_emb_kernel <<<S, KP, 0, stream>>>(sentence, embMat, embB);
  build_wcat_kernel<<<(NP1 * KP + 255) / 256, 256, 0, stream>>>(W_f, W_b, WcatT);
  build_wp_kernel  <<<(NP2 * KP + 255) / 256, 256, 0, stream>>>(W_p, WpT);
  zero_bf16_kernel <<<((S * KP) + 255) / 256, 256, 0, stream>>>(hiddenB, S * KP);

  // 2) Z = emb @ [W_f | W_b]   (WMMA bf16, f32 accumulate)
  {
    dim3 grid(NP1 / 16, S / 16);
    wmma_gemm_bf16<<<grid, 32, 0, stream>>>(embB, WcatT, Z, NP1);
  }

  // 3) sequential biLSTM scan (2 blocks: fwd, bwd)
  lstm_scan_kernel<<<2, 640, 0, stream>>>(Z, b_f, b_b, U_f, U_b, hidden, hiddenB);

  // 4) outP = hidden @ W_p   (WMMA bf16)
  {
    dim3 grid(NP2 / 16, S / 16);
    wmma_gemm_bf16<<<grid, 32, 0, stream>>>(hiddenB, WpT, outP, NP2);
  }

  // 5) primary + secondary attention, folded coeff*h_hat
  primary_attn_kernel<<<S, 128, 0, stream>>>(outP, b_p, hidden, synIdx, embMat, W_s, b_s, ch);

  // 6) deterministic reduction over S, then output heads
  reduce_stage1_kernel<<<CHUNKS, 640, 0, stream>>>(ch, partial);
  final_heads_kernel<<<1, 320, 0, stream>>>(partial, W_emo, b_emo, W_sent, b_sent, out);
}
